// Triaffine_45243185496531
// MI455X (gfx1250) — matrix operand order repaired and checked by reading, not compiled, not run on hardware
//
#include <hip/hip_runtime.h>

// Triaffine scoring s[b,z,x,y] = xb^T (z^T W) yb, B=4,S=128,D=512, n_out=1.
// 3-stage bf16 WMMA pipeline (f32 accumulate), weight streamed once from HBM.
//   P[k][x'][j] = sum_i xb[x',i] W[i,k,j]   (dominant ~146 GFLOP GEMM)
//   G[b][k][x*128+y] = P[k][b*128..][:] @ yb[b]^T
//   s[b][z][x*128+y] = z[b] @ G[b]          (fp32 contiguous output stores)

typedef __attribute__((ext_vector_type(16))) __bf16 v16bf;
typedef __attribute__((ext_vector_type(8)))  float  v8f;

namespace {
constexpr int Dm   = 512;           // n_in
constexpr int Dp1  = 513;           // n_in + bias
constexpr int JP   = 544;           // padded i/j contraction length (mult of 32)
constexpr int XP   = 512;           // B*S rows
constexpr int NXY  = 128 * 128;     // flattened (x,y)
constexpr size_t W_ISTRIDE = (size_t)Dm * Dp1;  // 262656 floats per i-row

constexpr size_t SZ_P  = (size_t)Dm * XP * JP * 2;       // 285,212,672 B
constexpr size_t SZ_G  = (size_t)4 * Dm * NXY * 2;       //  67,108,864 B
constexpr size_t SZ_AX = (size_t)XP * Dm * 2;            // xb  bf16 (no bias col; folded)
constexpr size_t SZ_YB = (size_t)XP * JP * 2;            // yb  bf16 padded (j==512 -> 1)
}

union FragU { uint4 q[2]; v16bf v; };

// A fragment (16x32, MxK) from row-major [M][K]; rows must be 16B aligned.
__device__ __forceinline__ v16bf ldA(const __bf16* __restrict__ base, int stride,
                                     int m0, int k0, int lane) {
  const int row = m0 + (lane & 15);
  const int kk  = k0 + ((lane >> 4) << 3);          // lanes 16-31: K chunks {8..15,24..31}
  const __bf16* p = base + (size_t)row * stride + kk;
  FragU f;
  f.q[0] = *reinterpret_cast<const uint4*>(p);       // K +0..7
  f.q[1] = *reinterpret_cast<const uint4*>(p + 16);  // K +16..23
  return f.v;
}

// B fragment (32x16, KxN) from row-major [N][K] (i.e. B^T); two 16B chunks per lane (K +0..7, +16..23).
// ISA layout: lanes 0-15 hold K={0..7,16..23} of column N=lane; lanes 16-31 hold K={8..15,24..31}.
__device__ __forceinline__ v16bf ldB(const __bf16* base, int stride,
                                     int n0, int k0, int lane) {
  const int row = n0 + (lane & 15);
  const int kk  = k0 + ((lane >> 4) << 3);
  const __bf16* p = base + (size_t)row * stride + kk;
  FragU f;
  f.q[0] = *reinterpret_cast<const uint4*>(p);
  f.q[1] = *reinterpret_cast<const uint4*>(p + 16);
  return f.v;
}

__device__ __forceinline__ v8f wmma_bf16(v16bf a, v16bf b, v8f c) {
  return __builtin_amdgcn_wmma_f32_16x16x32_bf16(false, a, false, b, (short)0, c,
                                                 false, false);
}

// ---------------- pack x/y/z to bf16 (yb padded: j==512 -> 1, j>512 -> 0) -------------
__global__ void kPack(const float* __restrict__ x, const float* __restrict__ y,
                      const float* __restrict__ z, __bf16* __restrict__ Ax,
                      __bf16* __restrict__ Yb, __bf16* __restrict__ Zb) {
  int i = blockIdx.x * blockDim.x + threadIdx.x;
  const int NX = XP * Dm, NY = XP * JP, NZ = XP * Dm;
  if (i < NX) { Ax[i] = (__bf16)x[i]; return; }
  i -= NX;
  if (i < NY) {
    const int row = i / JP, j = i % JP;
    float v = (j < Dm) ? y[row * Dm + j] : (j == Dm ? 1.f : 0.f);
    Yb[i] = (__bf16)v;
    return;
  }
  i -= NY;
  if (i < NZ) Zb[i] = (__bf16)z[i];
}

// ---------------- stage 1: P[k][x'][j] = sum_i xb W[i,k,j]  (+ bias row i=512) --------
// grid (512 k, 17 j-blocks of 32), 512 threads = 16 waves (8M x 2N); wave: 4 m-tiles.
// Weight slab [32i x 32j] staged fp32->bf16 transposed into double-buffered LDS [j][i];
// next slab's (branchless, address-clamped) loads issued before current WMMAs.
__global__ __launch_bounds__(512) void kA(const float* __restrict__ W,
                                          const __bf16* __restrict__ Ax,
                                          __bf16* __restrict__ P) {
  const int k  = blockIdx.x;
  const int j0 = blockIdx.y * 32;
  const int t = threadIdx.x, lane = t & 31, w = t >> 5;
  const int wn = w & 1, m0 = (w >> 1) * 64;
  __shared__ __align__(16) __bf16 sB[2][32][40];  // [buf][j][i], pitch 40

  v8f acc[4];
  const v8f zero = {0.f,0.f,0.f,0.f,0.f,0.f,0.f,0.f};
  acc[0] = zero; acc[1] = zero; acc[2] = zero; acc[3] = zero;

  const int fi = t >> 4;       // 0..31 : i row within step
  const int fq = t & 15;       // j pair
  const size_t wk = (size_t)k * Dp1;
  const int ja = j0 + 2 * fq;
  const int jb = ja + 1;
  // clamped, unconditional load addresses (value zeroed by select when OOB)
  const size_t ca = wk + (ja < Dp1 ? ja : Dp1 - 1);
  const size_t cb = wk + (jb < Dp1 ? jb : Dp1 - 1);
  const float ma = (ja < Dp1) ? 1.f : 0.f;
  const float mb = (jb < Dp1) ? 1.f : 0.f;

  // prologue: slab for it=0 into buffer 0
  {
    const size_t rowbase = (size_t)fi * W_ISTRIDE;
    sB[0][2 * fq][fi]     = (__bf16)(W[rowbase + ca] * ma);
    sB[0][2 * fq + 1][fi] = (__bf16)(W[rowbase + cb] * mb);
  }
  __syncthreads();

  for (int it = 0; it < 16; ++it) {
    const int cur = it & 1, nxt = cur ^ 1;
    // issue next slab's loads first so they overlap the WMMAs below
    float n0 = 0.f, n1 = 0.f;
    if (it < 15) {
      const size_t rowbase = (size_t)((it + 1) * 32 + fi) * W_ISTRIDE;
      n0 = W[rowbase + ca] * ma;
      n1 = W[rowbase + cb] * mb;
    }
    const int kk = it * 32;
    const v16bf b = ldB(&sB[cur][0][0], 40, wn * 16, 0, lane);
#pragma unroll
    for (int mt = 0; mt < 4; ++mt) {
      const v16bf a = ldA(Ax, Dm, m0 + mt * 16, kk, lane);
      acc[mt] = wmma_bf16(a, b, acc[mt]);
    }
    if (it < 15) {
      sB[nxt][2 * fq][fi]     = (__bf16)n0;
      sB[nxt][2 * fq + 1][fi] = (__bf16)n1;
    }
    __syncthreads();
  }
  // xb bias column (i=512, value 1): add W[512,k,j] to every output row.
  {
    const int j = j0 + wn * 16 + (lane & 15);
    const int jc = (j < Dp1) ? j : Dp1 - 1;
    const float bias = W[(size_t)Dm * W_ISTRIDE + wk + jc] * ((j < Dp1) ? 1.f : 0.f);
#pragma unroll
    for (int mt = 0; mt < 4; ++mt)
#pragma unroll
      for (int r = 0; r < 8; ++r) acc[mt][r] += bias;
  }
  __bf16* out = P + (size_t)k * XP * JP;
  const int c = j0 + wn * 16 + (lane & 15);
#pragma unroll
  for (int mt = 0; mt < 4; ++mt) {
    const int rb = m0 + mt * 16 + ((lane >> 4) << 3);
#pragma unroll
    for (int r = 0; r < 8; ++r)
      out[(size_t)(rb + r) * JP + c] = (__bf16)acc[mt][r];
  }
}

// ---------------- stage 2: G[b][k][x*128+y] = P[k][b-slice] @ yb[b]^T ------------------
// grid (512 k, 4 b), 512 threads = 16 waves (4x4); wave tile 32x32. K = JP = 544.
__global__ __launch_bounds__(512) void kB(const __bf16* __restrict__ P,
                                          const __bf16* __restrict__ Yb,
                                          __bf16* __restrict__ G) {
  const int k = blockIdx.x, b = blockIdx.y;
  const int t = threadIdx.x, lane = t & 31, w = t >> 5;
  const int m0 = (w & 3) * 32, n0 = (w >> 2) * 32;
  const __bf16* A  = P  + ((size_t)k * XP + b * 128) * JP;
  const __bf16* Bm = Yb + (size_t)b * 128 * JP;

  v8f acc[2][2];
  const v8f zero = {0.f,0.f,0.f,0.f,0.f,0.f,0.f,0.f};
  acc[0][0] = zero; acc[0][1] = zero; acc[1][0] = zero; acc[1][1] = zero;

  for (int kk = 0; kk < JP; kk += 32) {
    const v16bf a0 = ldA(A, JP, m0,      kk, lane);
    const v16bf a1 = ldA(A, JP, m0 + 16, kk, lane);
    const v16bf b0 = ldB(Bm, JP, n0,      kk, lane);
    const v16bf b1 = ldB(Bm, JP, n0 + 16, kk, lane);
    acc[0][0] = wmma_bf16(a0, b0, acc[0][0]);
    acc[0][1] = wmma_bf16(a0, b1, acc[0][1]);
    acc[1][0] = wmma_bf16(a1, b0, acc[1][0]);
    acc[1][1] = wmma_bf16(a1, b1, acc[1][1]);
  }
  __bf16* out = G + ((size_t)b * Dm + k) * NXY;
#pragma unroll
  for (int mt = 0; mt < 2; ++mt) {
    const int rb = m0 + mt * 16 + ((lane >> 4) << 3);
#pragma unroll
    for (int nt = 0; nt < 2; ++nt) {
      const int c = n0 + nt * 16 + (lane & 15);
#pragma unroll
      for (int r = 0; r < 8; ++r)
        out[(size_t)(rb + r) * 128 + c] = (__bf16)acc[mt][nt][r];
    }
  }
}

// ---------------- stage 3: s[b][z][n] = z[b] @ G[b]  (n = x*128+y) --------------------
// grid (64 n-blocks of 256, 4 b), 512 threads = 16 waves (4x4); wave 32M x 64N.
// G is [K][N]; 32x256 slab transposed through double-buffered LDS -> [n][k].
__global__ __launch_bounds__(512) void kC(const __bf16* __restrict__ Zb,
                                          const __bf16* __restrict__ G,
                                          float* __restrict__ out) {
  const int b = blockIdx.y;
  const int n0 = blockIdx.x * 256;
  const int t = threadIdx.x, lane = t & 31, w = t >> 5;
  const int m0 = (w & 3) * 32, nl0 = (w >> 2) * 64;
  __shared__ __align__(16) __bf16 sG[2][256][40];   // [buf][n][k], pitch 40

  const __bf16* A  = Zb + (size_t)b * 128 * Dm;
  const __bf16* Gb = G  + (size_t)b * Dm * NXY;

  v8f acc[2][4];
  const v8f zero = {0.f,0.f,0.f,0.f,0.f,0.f,0.f,0.f};
#pragma unroll
  for (int mt = 0; mt < 2; ++mt)
#pragma unroll
    for (int nt = 0; nt < 4; ++nt) acc[mt][nt] = zero;

  const int fq = t >> 5;   // 0..15 : 16-col n-chunk
  const int fr = t & 31;   // k row
  const __bf16* srcBase = Gb + (size_t)fr * NXY + n0 + fq * 16;

  // prologue: slab k0=0 into buffer 0
  {
    FragU f;
    f.q[0] = *reinterpret_cast<const uint4*>(srcBase);
    f.q[1] = *reinterpret_cast<const uint4*>(srcBase + 8);
#pragma unroll
    for (int e = 0; e < 16; ++e) sG[0][fq * 16 + e][fr] = f.v[e];
  }
  __syncthreads();

  for (int it = 0; it < 16; ++it) {
    const int cur = it & 1, nxt = cur ^ 1;
    const int k0 = it * 32;
    FragU fn;
    if (it < 15) {
      const __bf16* src = srcBase + (size_t)(k0 + 32) * NXY;
      fn.q[0] = *reinterpret_cast<const uint4*>(src);
      fn.q[1] = *reinterpret_cast<const uint4*>(src + 8);
    }
    const v16bf a0 = ldA(A, Dm, m0,      k0, lane);
    const v16bf a1 = ldA(A, Dm, m0 + 16, k0, lane);
#pragma unroll
    for (int nt = 0; nt < 4; ++nt) {
      const v16bf bb = ldB(&sG[cur][0][0], 40, nl0 + nt * 16, 0, lane);
      acc[0][nt] = wmma_bf16(a0, bb, acc[0][nt]);
      acc[1][nt] = wmma_bf16(a1, bb, acc[1][nt]);
    }
    if (it < 15) {
#pragma unroll
      for (int e = 0; e < 16; ++e) sG[nxt][fq * 16 + e][fr] = fn.v[e];
    }
    __syncthreads();
  }
  float* ob = out + (size_t)b * (128 * NXY);
#pragma unroll
  for (int mt = 0; mt < 2; ++mt) {
    const int rb = m0 + mt * 16 + ((lane >> 4) << 3);
#pragma unroll
    for (int nt = 0; nt < 4; ++nt) {
      const int c = n0 + nl0 + nt * 16 + (lane & 15);
#pragma unroll
      for (int r = 0; r < 8; ++r)
        ob[(size_t)(rb + r) * NXY + c] = acc[mt][nt][r];
    }
  }
}

extern "C" void kernel_launch(void* const* d_in, const int* in_sizes, int n_in,
                              void* d_out, int out_size, void* d_ws, size_t ws_size,
                              hipStream_t stream) {
  const float* x = (const float*)d_in[0];
  const float* y = (const float*)d_in[1];
  const float* z = (const float*)d_in[2];
  const float* w = (const float*)d_in[3];
  float* out = (float*)d_out;

  char* ws = (char*)d_ws;        // needs ~354 MB of scratch
  __bf16* P  = (__bf16*)(ws);
  __bf16* G  = (__bf16*)(ws + SZ_P);
  __bf16* Ax = (__bf16*)(ws + SZ_P + SZ_G);
  __bf16* Yb = (__bf16*)(ws + SZ_P + SZ_G + SZ_AX);
  __bf16* Zb = (__bf16*)(ws + SZ_P + SZ_G + SZ_AX + SZ_YB);

  {
    const int total = XP * Dm + XP * JP + XP * Dm;
    kPack<<<(total + 255) / 256, 256, 0, stream>>>(x, y, z, Ax, Yb, Zb);
  }
  {
    dim3 g(Dm, JP / 32);          // (512, 17)
    kA<<<g, 512, 0, stream>>>(w, Ax, P);
  }
  {
    dim3 g(Dm, 4);                // (512, 4)
    kB<<<g, 512, 0, stream>>>(P, Yb, G);
  }
  {
    dim3 g(NXY / 256, 4);         // (64, 4)
    kC<<<g, 512, 0, stream>>>(Zb, G, out);
  }
}
